// Block_89816356094621
// MI455X (gfx1250) — matrix operand order repaired and checked by reading, not compiled, not run on hardware
//
#include <hip/hip_runtime.h>
#include <hip/hip_bf16.h>

typedef __bf16 bf16;
typedef __attribute__((ext_vector_type(16))) __bf16 v16bf;
typedef __attribute__((ext_vector_type(8)))  __bf16 v8bf;
typedef __attribute__((ext_vector_type(8)))  float  v8f;

static constexpr int BB = 64, TT = 256, CC = 512, HH = 8;
static constexpr int MROWS = BB * TT;   // 16384
static constexpr int HC = HH * CC;      // 4096
static constexpr int FF = 4 * CC;       // 2048
static constexpr float NEG_INF = -3.0e38f;

// LDS tile leading dims (padded: breaks bank conflicts, keeps 32B row alignment)
static constexpr int LDA_S = 48;   // A tile rows of 32 bf16 + 16 pad
static constexpr int LDB_S = 144;  // B tile rows of 128 bf16 + 16 pad

union ABFrag { v16bf v; v8bf h[2]; };

// A fragment 16x32 bf16 (row-major source, lda in elements).
// lane 0-15: M=lane, K = k0+0..7 and k0+16..23 ; lane 16-31: M=lane-16, K = k0+8..15 / k0+24..31
__device__ __forceinline__ v16bf load_frag_a(const bf16* A, int lda, int lane) {
  const bf16* p = A + (size_t)(lane & 15) * lda + ((lane >> 4) << 3);
  ABFrag u;
  u.h[0] = *(const v8bf*)(p);
  u.h[1] = *(const v8bf*)(p + 16);
  return u.v;
}

// B fragment 32x16 bf16 (row-major source): lane holds column N=lane%16, K = k0+8h+0..7 and k0+16+8h+0..7 (h=lane>>4)
__device__ __forceinline__ v16bf load_frag_b(const bf16* Bm, int ldb, int lane) {
  const bf16* p = Bm + (lane & 15) + (size_t)((lane >> 4) << 3) * ldb;
  v16bf b;
#pragma unroll
  for (int s = 0; s < 8; ++s) {
    b[s] = p[(size_t)s * ldb];
    b[8 + s] = p[(size_t)(16 + s) * ldb];
  }
  return b;
}

__device__ __forceinline__ v8f wmma_bf16(v16bf a, v16bf b, v8f c) {
  return __builtin_amdgcn_wmma_f32_16x16x32_bf16(false, a, false, b, (short)0, c, false, false);
}

// ---- CDNA5 async global->LDS (ASYNCcnt) ----
__device__ __forceinline__ void async_copy16(const bf16* g, bf16* l) {
  unsigned loff = (unsigned)(uintptr_t)l;  // low 32 bits of LDS aperture addr = LDS offset
  asm volatile("global_load_async_to_lds_b128 %0, %1, off" :: "v"(loff), "v"(g) : "memory");
}
#define ASYNC_WAIT_4() asm volatile("s_wait_asynccnt 4" ::: "memory")
#define ASYNC_WAIT_0() asm volatile("s_wait_asynccnt 0" ::: "memory")

struct GemmSmem {
  __align__(32) bf16 A[2][128 * LDA_S];  // 2 x 12KB
  __align__(32) bf16 B[2][32 * LDB_S];   // 2 x 9KB
};

// Block cooperative stage load: A tile 128x32 (from Ag, lda), B tile 32x128 (from Bg, ldb).
// 4 async b128 ops per thread-slot => 4 outstanding per wave per stage.
__device__ __forceinline__ void stage_load(const bf16* Ag, int lda, const bf16* Bg, int ldb,
                                           bf16* sA, bf16* sB, int tid) {
#pragma unroll
  for (int c = 0; c < 2; ++c) {               // A: 512 chunks of 8 elems
    int ch = tid * 2 + c;
    int row = ch >> 2, sub = ch & 3;
    async_copy16(Ag + (size_t)row * lda + sub * 8, sA + row * LDA_S + sub * 8);
  }
#pragma unroll
  for (int c = 0; c < 2; ++c) {               // B: 512 chunks of 8 elems
    int ch = tid * 2 + c;
    int row = ch >> 4, sub = ch & 15;
    async_copy16(Bg + (size_t)row * ldb + sub * 8, sB + row * LDB_S + sub * 8);
  }
}

// Block 128x128 GEMM with double-buffered async-LDS pipeline.
// Wave wid: wm=wid>>2 (0..1) -> 64 rows, wn=wid&3 (0..3) -> 32 cols. acc = 4x2 16x16 tiles.
__device__ __forceinline__ void gemm_block(const bf16* Ag, int lda, const bf16* Bg, int ldb,
                                           int K, GemmSmem& sm, v8f (&acc)[4][2]) {
  int tid = threadIdx.x;
  int wid = tid >> 5, lane = tid & 31;
  int wm = wid >> 2, wn = wid & 3;
  int nk = K >> 5;

  stage_load(Ag, lda, Bg, ldb, sm.A[0], sm.B[0], tid);
  for (int kk = 0; kk < nk; ++kk) {
    int cur = kk & 1;
    if (kk + 1 < nk) {
      stage_load(Ag + (size_t)(kk + 1) * 32, lda, Bg + (size_t)(kk + 1) * 32 * ldb, ldb,
                 sm.A[cur ^ 1], sm.B[cur ^ 1], tid);
      ASYNC_WAIT_4();   // stage kk complete; stage kk+1 (4 ops) still in flight
    } else {
      ASYNC_WAIT_0();
    }
    __syncthreads();    // all waves' stage-kk data visible in LDS

    const bf16* sA = sm.A[cur] + (size_t)(wm * 64) * LDA_S;
    const bf16* sB = sm.B[cur] + wn * 32;
    v16bf b0 = load_frag_b(sB, LDB_S, lane);
    v16bf b1 = load_frag_b(sB + 16, LDB_S, lane);
#pragma unroll
    for (int i = 0; i < 4; ++i) {
      v16bf a = load_frag_a(sA + (size_t)(i * 16) * LDA_S, LDA_S, lane);
      acc[i][0] = wmma_bf16(a, b0, acc[i][0]);
      acc[i][1] = wmma_bf16(a, b1, acc[i][1]);
    }
    __syncthreads();    // done reading buf `cur` before stage kk+2 overwrites it
  }
}

// ---------------- fp32 -> bf16 convert ----------------
__global__ __launch_bounds__(256) void cvt_kernel(const float* __restrict__ src,
                                                  bf16* __restrict__ dst, int n) {
  int i = blockIdx.x * 256 + threadIdx.x;
  if (i < n) dst[i] = (bf16)src[i];
}

// ---------------- QKV projections (24 batched GEMMs 16384x512x512) ----------------
__global__ __launch_bounds__(256) void qkv_gemm(const bf16* __restrict__ xb,
                                                const bf16* __restrict__ wqkv,
                                                const float* __restrict__ Kb,
                                                const float* __restrict__ Qb,
                                                const float* __restrict__ Vb,
                                                bf16* __restrict__ q, bf16* __restrict__ kT,
                                                bf16* __restrict__ v) {
  __shared__ GemmSmem sm;
  int z = blockIdx.z, p = z >> 3, hh = z & 7;
  int wid = threadIdx.x >> 5, lane = threadIdx.x & 31;
  int lh = lane >> 4, ln = lane & 15;
  int mb = blockIdx.y * 128 + (wid >> 2) * 64;
  int nb = blockIdx.x * 128 + (wid & 3) * 32;

  v8f acc[4][2] = {};
  gemm_block(xb + (size_t)(blockIdx.y * 128) * CC, CC,
             wqkv + (size_t)z * CC * CC + blockIdx.x * 128, CC, CC, sm, acc);

  const float* bias = ((p == 0) ? Kb : (p == 1) ? Qb : Vb) + hh * CC;
  if (p == 0) {
    // transposed store [H,B,C,T]: lane's 8 acc rows are consecutive t for fixed d -> 16B store
#pragma unroll
    for (int i = 0; i < 4; ++i)
#pragma unroll
      for (int j = 0; j < 2; ++j) {
        int d = nb + j * 16 + ln;
        int Mb0 = mb + i * 16 + lh * 8;
        int bat = Mb0 >> 8, t0 = Mb0 & (TT - 1);
        float bs = bias[d];
        v8bf pk;
#pragma unroll
        for (int r = 0; r < 8; ++r) pk[r] = (bf16)(acc[i][j][r] + bs);
        *(v8bf*)(kT + (((size_t)hh * BB + bat) * CC + d) * TT + t0) = pk;
      }
  } else {
    bf16* out = (p == 1) ? q : v;
#pragma unroll
    for (int i = 0; i < 4; ++i)
#pragma unroll
      for (int j = 0; j < 2; ++j) {
        int d = nb + j * 16 + ln;
        float bs = bias[d];
#pragma unroll
        for (int r = 0; r < 8; ++r) {
          int M = mb + i * 16 + lh * 8 + r;
          int bat = M >> 8, t = M & (TT - 1);
          out[(((size_t)hh * BB + bat) * TT + t) * CC + d] = (bf16)(acc[i][j][r] + bs);
        }
      }
  }
}

// ---------------- fused attention: one block per (b,h,32-row t-stripe) ----------------
__global__ __launch_bounds__(256) void attn_kernel(const bf16* __restrict__ q,
                                                   const bf16* __restrict__ kT,
                                                   const bf16* __restrict__ v,
                                                   bf16* __restrict__ cat) {
  __shared__ __align__(32) float sc[32][TT];
  __shared__ __align__(32) bf16  pr[32][TT];
  __shared__ float red[32][8];
  __shared__ float rowmax[32], rowsum[32];

  int stripe = blockIdx.x, hh = blockIdx.y, bat = blockIdx.z;
  int t0 = stripe * 32;
  int wid = threadIdx.x >> 5, lane = threadIdx.x & 31;
  int lh = lane >> 4, ln = lane & 15;

  const bf16* qh = q  + ((size_t)hh * BB + bat) * TT * CC;  // [T][C]
  const bf16* kh = kT + ((size_t)hh * BB + bat) * CC * TT;  // [C][T]
  const bf16* vh = v  + ((size_t)hh * BB + bat) * TT * CC;  // [T][C]

  // phase 1: scores[32 x 256]; wave handles 32-col slice s0
  int s0 = wid * 32;
  {
    v8f acc[2][2] = {};
    for (int k0 = 0; k0 < CC; k0 += 32) {
      v16bf b0 = load_frag_b(kh + (size_t)k0 * TT + s0, TT, lane);
      v16bf b1 = load_frag_b(kh + (size_t)k0 * TT + s0 + 16, TT, lane);
#pragma unroll
      for (int i = 0; i < 2; ++i) {
        v16bf a = load_frag_a(qh + (size_t)(t0 + i * 16) * CC + k0, CC, lane);
        acc[i][0] = wmma_bf16(a, b0, acc[i][0]);
        acc[i][1] = wmma_bf16(a, b1, acc[i][1]);
      }
    }
    const float scale = 0.04419417382415922f;  // 1/sqrt(512)
#pragma unroll
    for (int i = 0; i < 2; ++i)
#pragma unroll
      for (int j = 0; j < 2; ++j)
#pragma unroll
        for (int r = 0; r < 8; ++r) {
          int tl = i * 16 + lh * 8 + r;
          int s = s0 + j * 16 + ln;
          sc[tl][s] = (s <= t0 + tl) ? acc[i][j][r] * scale : NEG_INF;
        }
  }
  __syncthreads();

  // phase 2: row softmax (8 threads per row, 32 cols each)
  {
    int row = threadIdx.x >> 3, seg = threadIdx.x & 7;
    float m = NEG_INF;
    for (int c = seg * 32; c < seg * 32 + 32; ++c) m = fmaxf(m, sc[row][c]);
    red[row][seg] = m;
    __syncthreads();
    if (seg == 0) {
      float mm = red[row][0];
#pragma unroll
      for (int k = 1; k < 8; ++k) mm = fmaxf(mm, red[row][k]);
      rowmax[row] = mm;
    }
    __syncthreads();
    float mm = rowmax[row], s = 0.0f;
    for (int c = seg * 32; c < seg * 32 + 32; ++c) {
      float e = __expf(sc[row][c] - mm);
      sc[row][c] = e;
      s += e;
    }
    red[row][seg] = s;
    __syncthreads();
    if (seg == 0) {
      float ss = red[row][0];
#pragma unroll
      for (int k = 1; k < 8; ++k) ss += red[row][k];
      rowsum[row] = ss;
    }
    __syncthreads();
    float inv = 1.0f / rowsum[row];
    for (int c = seg * 32; c < seg * 32 + 32; ++c) pr[row][c] = (bf16)(sc[row][c] * inv);
  }
  __syncthreads();

  // phase 3: out[32 x 512] = probs @ v; wave handles 64-col d-slice; causal-truncated K loop
  {
    int d0 = wid * 64;
    v8f oacc[2][4] = {};
    const bf16* prp = &pr[0][0];
    int nk = stripe + 1;
    for (int kk = 0; kk < nk; ++kk) {
      int k0 = kk * 32;
      v16bf bf[4];
#pragma unroll
      for (int j = 0; j < 4; ++j)
        bf[j] = load_frag_b(vh + (size_t)k0 * CC + d0 + j * 16, CC, lane);
#pragma unroll
      for (int i = 0; i < 2; ++i) {
        v16bf a = load_frag_a(prp + (size_t)(i * 16) * TT + k0, TT, lane);
#pragma unroll
        for (int j = 0; j < 4; ++j) oacc[i][j] = wmma_bf16(a, bf[j], oacc[i][j]);
      }
    }
#pragma unroll
    for (int i = 0; i < 2; ++i)
#pragma unroll
      for (int j = 0; j < 4; ++j)
#pragma unroll
        for (int r = 0; r < 8; ++r) {
          int t = t0 + i * 16 + lh * 8 + r;
          int d = d0 + j * 16 + ln;
          cat[(size_t)(bat * TT + t) * HC + hh * CC + d] = (bf16)oacc[i][j][r];
        }
  }
}

// ---------------- output projection: r1 = cat @ Pw + Pb + x (f32) ----------------
__global__ __launch_bounds__(256) void proj_gemm(const bf16* __restrict__ cat,
                                                 const bf16* __restrict__ Pwb,
                                                 const float* __restrict__ Pb,
                                                 const float* __restrict__ x,
                                                 float* __restrict__ r1) {
  __shared__ GemmSmem sm;
  int wid = threadIdx.x >> 5, lane = threadIdx.x & 31;
  int lh = lane >> 4, ln = lane & 15;
  int mb = blockIdx.y * 128 + (wid >> 2) * 64;
  int nb = blockIdx.x * 128 + (wid & 3) * 32;
  v8f acc[4][2] = {};
  gemm_block(cat + (size_t)(blockIdx.y * 128) * HC, HC, Pwb + blockIdx.x * 128, CC, HC, sm, acc);
#pragma unroll
  for (int i = 0; i < 4; ++i)
#pragma unroll
    for (int j = 0; j < 2; ++j) {
      int n = nb + j * 16 + ln;
      float bs = Pb[n];
#pragma unroll
      for (int r = 0; r < 8; ++r) {
        int M = mb + i * 16 + lh * 8 + r;
        size_t idx = (size_t)M * CC + n;
        r1[idx] = acc[i][j][r] + bs + x[idx];
      }
    }
}

// ---------------- layernorm: one wave per 512-elem row ----------------
__global__ __launch_bounds__(256) void ln_kernel(const float* __restrict__ in,
                                                 const float* __restrict__ g,
                                                 const float* __restrict__ be,
                                                 float* __restrict__ outf,
                                                 bf16* __restrict__ outb) {
  int row = blockIdx.x * 8 + (threadIdx.x >> 5);
  int lane = threadIdx.x & 31;
  const float* p = in + (size_t)row * CC;
  float vals[16];
  float s = 0.0f;
#pragma unroll
  for (int e = 0; e < 16; ++e) { vals[e] = p[lane + e * 32]; s += vals[e]; }
#pragma unroll
  for (int off = 16; off > 0; off >>= 1) s += __shfl_xor(s, off, 32);
  float mu = s * (1.0f / CC);
  float vs = 0.0f;
#pragma unroll
  for (int e = 0; e < 16; ++e) { float d = vals[e] - mu; vs += d * d; }
#pragma unroll
  for (int off = 16; off > 0; off >>= 1) vs += __shfl_xor(vs, off, 32);
  float rs = rsqrtf(vs * (1.0f / CC) + 1e-5f);
#pragma unroll
  for (int e = 0; e < 16; ++e) {
    int c = lane + e * 32;
    float y = (vals[e] - mu) * rs * g[c] + be[c];
    if (outf) outf[(size_t)row * CC + c] = y;
    if (outb) outb[(size_t)row * CC + c] = (bf16)y;
  }
}

// ---------------- FFN1: relu(o1 @ W1 + b1) -> bf16 ----------------
__global__ __launch_bounds__(256) void ffn1_gemm(const bf16* __restrict__ o1b,
                                                 const bf16* __restrict__ W1b,
                                                 const float* __restrict__ b1,
                                                 bf16* __restrict__ ffh) {
  __shared__ GemmSmem sm;
  int wid = threadIdx.x >> 5, lane = threadIdx.x & 31;
  int lh = lane >> 4, ln = lane & 15;
  int mb = blockIdx.y * 128 + (wid >> 2) * 64;
  int nb = blockIdx.x * 128 + (wid & 3) * 32;
  v8f acc[4][2] = {};
  gemm_block(o1b + (size_t)(blockIdx.y * 128) * CC, CC, W1b + blockIdx.x * 128, FF, CC, sm, acc);
#pragma unroll
  for (int i = 0; i < 4; ++i)
#pragma unroll
    for (int j = 0; j < 2; ++j) {
      int n = nb + j * 16 + ln;
      float bs = b1[n];
#pragma unroll
      for (int r = 0; r < 8; ++r) {
        int M = mb + i * 16 + lh * 8 + r;
        float y = acc[i][j][r] + bs;
        ffh[(size_t)M * FF + n] = (bf16)(y > 0.0f ? y : 0.0f);
      }
    }
}

// ---------------- FFN2: r2 = ffh @ W2 + b2 + o1 (f32) ----------------
__global__ __launch_bounds__(256) void ffn2_gemm(const bf16* __restrict__ ffh,
                                                 const bf16* __restrict__ W2b,
                                                 const float* __restrict__ b2,
                                                 const float* __restrict__ o1f,
                                                 float* __restrict__ r2) {
  __shared__ GemmSmem sm;
  int wid = threadIdx.x >> 5, lane = threadIdx.x & 31;
  int lh = lane >> 4, ln = lane & 15;
  int mb = blockIdx.y * 128 + (wid >> 2) * 64;
  int nb = blockIdx.x * 128 + (wid & 3) * 32;
  v8f acc[4][2] = {};
  gemm_block(ffh + (size_t)(blockIdx.y * 128) * FF, FF, W2b + blockIdx.x * 128, CC, FF, sm, acc);
#pragma unroll
  for (int i = 0; i < 4; ++i)
#pragma unroll
    for (int j = 0; j < 2; ++j) {
      int n = nb + j * 16 + ln;
      float bs = b2[n];
#pragma unroll
      for (int r = 0; r < 8; ++r) {
        int M = mb + i * 16 + lh * 8 + r;
        size_t idx = (size_t)M * CC + n;
        r2[idx] = acc[i][j][r] + bs + o1f[idx];
      }
    }
}

extern "C" void kernel_launch(void* const* d_in, const int* in_sizes, int n_in,
                              void* d_out, int out_size, void* d_ws, size_t ws_size,
                              hipStream_t stream) {
  (void)in_sizes; (void)n_in; (void)out_size; (void)ws_size;
  char* ws = (char*)d_ws;
  size_t off = 0;
  auto take = [&](size_t bytes) -> void* {
    off = (off + 255) & ~(size_t)255;
    void* p = ws + off;
    off += bytes;
    return p;
  };

  bf16* xb   = (bf16*)take((size_t)MROWS * CC * 2);
  bf16* wqkv = (bf16*)take((size_t)3 * HH * CC * CC * 2);
  bf16* Pwb  = (bf16*)take((size_t)HC * CC * 2);
  bf16* W1b  = (bf16*)take((size_t)CC * FF * 2);
  bf16* W2b  = (bf16*)take((size_t)FF * CC * 2);
  bf16* qb   = (bf16*)take((size_t)HH * BB * TT * CC * 2);
  bf16* kTb  = (bf16*)take((size_t)HH * BB * CC * TT * 2);
  bf16* vb   = (bf16*)take((size_t)HH * BB * TT * CC * 2);
  bf16* cat  = (bf16*)take((size_t)MROWS * HC * 2);
  float* r1  = (float*)take((size_t)MROWS * CC * 4);
  float* o1f = (float*)take((size_t)MROWS * CC * 4);
  bf16* o1b  = (bf16*)take((size_t)MROWS * CC * 2);
  bf16* ffh  = (bf16*)take((size_t)MROWS * FF * 2);
  float* r2  = (float*)take((size_t)MROWS * CC * 4);

  auto cvt = [&](const void* src, bf16* dst, int n) {
    cvt_kernel<<<dim3((n + 255) / 256), dim3(256), 0, stream>>>((const float*)src, dst, n);
  };
  cvt(d_in[0], xb, MROWS * CC);
  cvt(d_in[1], wqkv, HH * CC * CC);
  cvt(d_in[3], wqkv + (size_t)HH * CC * CC, HH * CC * CC);
  cvt(d_in[5], wqkv + (size_t)2 * HH * CC * CC, HH * CC * CC);
  cvt(d_in[7], Pwb, HC * CC);
  cvt(d_in[9], W1b, CC * FF);
  cvt(d_in[11], W2b, FF * CC);

  qkv_gemm<<<dim3(CC / 128, MROWS / 128, 24), dim3(256), 0, stream>>>(
      xb, wqkv, (const float*)d_in[2], (const float*)d_in[4], (const float*)d_in[6],
      qb, kTb, vb);

  attn_kernel<<<dim3(TT / 32, HH, BB), dim3(256), 0, stream>>>(qb, kTb, vb, cat);

  proj_gemm<<<dim3(CC / 128, MROWS / 128), dim3(256), 0, stream>>>(
      cat, Pwb, (const float*)d_in[8], (const float*)d_in[0], r1);

  ln_kernel<<<dim3(MROWS / 8), dim3(256), 0, stream>>>(
      r1, (const float*)d_in[13], (const float*)d_in[14], o1f, o1b);

  ffn1_gemm<<<dim3(FF / 128, MROWS / 128), dim3(256), 0, stream>>>(
      o1b, W1b, (const float*)d_in[10], ffh);

  ffn2_gemm<<<dim3(CC / 128, MROWS / 128), dim3(256), 0, stream>>>(
      ffh, W2b, (const float*)d_in[12], o1f, r2);

  ln_kernel<<<dim3(MROWS / 8), dim3(256), 0, stream>>>(
      r2, (const float*)d_in[15], (const float*)d_in[16], (float*)d_out, nullptr);
}
